// GlassboxKAN_11218454577798
// MI455X (gfx1250) — matrix operand order repaired and checked by reading, not compiled, not run on hardware
//
#include <hip/hip_runtime.h>
#include <hip/hip_bf16.h>
#include <cstddef>
#include <cstdint>

// ---------------------------------------------------------------------------
// GlassboxKAN forward for MI455X (gfx1250, wave32, WMMA).
//   out[b] = silu(silu(silu(x)@W1^T + b1)@W2^T + b2)@W3^T + b3 + x^T triu(Wprod) x
// Dominant cost: 8192x4096x4096 GEMM -> v_wmma_f32_16x16x32_f16, f32 accum,
// fed by double-buffered GLOBAL_LOAD_ASYNC_TO_LDS_B128 (ASYNCcnt) copies.
// K-loop unrolled by 2 so both LDS buffers are compile-time constants; all
// staging addresses hoisted, copies issued unconditionally (no exec masking).
// ---------------------------------------------------------------------------

#define BB 8192
#define DD 64
#define HH 4096

typedef __attribute__((ext_vector_type(16))) _Float16 v16h;
typedef __attribute__((ext_vector_type(8)))  float    v8f;
typedef __attribute__((ext_vector_type(2)))  _Float16 h2;

__device__ __forceinline__ float silu_f(float h) {
    return h / (1.0f + __expf(-h));
}

// addrspace(3) offsets: generic LDS pointer low 32 bits == workgroup LDS offset
__device__ __forceinline__ unsigned lds_off32(const void* p) {
    return (unsigned)(uintptr_t)p;
}

// CDNA5 async global->LDS copy, 16B per lane, tracked by ASYNCcnt.
__device__ __forceinline__ void async_copy_b128(unsigned lds_addr, const void* gaddr) {
    asm volatile("global_load_async_to_lds_b128 %0, %1, off"
                 :: "v"(lds_addr), "v"(gaddr) : "memory");
}

__device__ __forceinline__ void wait_async0() {
    asm volatile("s_wait_asynccnt 0x0" ::: "memory");
}

// ---------------------------------------------------------------------------
// Kernel 1: per-row quadratic form (upper-tri) + b3, and f16(silu(x)).
// One wave (32 lanes) per batch row. Initializes d_out.
// ---------------------------------------------------------------------------
__global__ __launch_bounds__(256)
void prep_kernel(const float* __restrict__ x, const float* __restrict__ Wprod,
                 const float* __restrict__ b3, _Float16* __restrict__ xs,
                 float* __restrict__ out) {
    __shared__ float xrow[8][DD];
    const int wid  = threadIdx.x >> 5;
    const int lane = threadIdx.x & 31;
    const int b    = blockIdx.x * 8 + wid;

    const float x0 = x[(size_t)b * DD + lane];
    const float x1 = x[(size_t)b * DD + 32 + lane];
    xrow[wid][lane]      = x0;
    xrow[wid][lane + 32] = x1;
    xs[(size_t)b * DD + lane]      = (_Float16)silu_f(x0);
    xs[(size_t)b * DD + 32 + lane] = (_Float16)silu_f(x1);
    __syncthreads();

    float partial = 0.0f;
    #pragma unroll
    for (int ii = 0; ii < 2; ++ii) {
        const int i  = lane + 32 * ii;
        const float xi = xrow[wid][i];
        for (int j = i; j < DD; ++j)
            partial += Wprod[i * DD + j] * xi * xrow[wid][j];
    }
    #pragma unroll
    for (int ofs = 16; ofs >= 1; ofs >>= 1)
        partial += __shfl_xor(partial, ofs, 32);
    if (lane == 0) out[b] = partial + b3[0];
}

// ---------------------------------------------------------------------------
// Kernel 2: f32 -> f16 weight conversion (count divisible by 4).
// ---------------------------------------------------------------------------
__global__ __launch_bounds__(256)
void cvt_kernel(const float* __restrict__ src, _Float16* __restrict__ dst, int n4) {
    const int i = blockIdx.x * blockDim.x + threadIdx.x;
    if (i < n4) {
        const float4 v = *(const float4*)(src + (size_t)i * 4);
        _Float16* d = dst + (size_t)i * 4;
        d[0] = (_Float16)v.x; d[1] = (_Float16)v.y;
        d[2] = (_Float16)v.z; d[3] = (_Float16)v.w;
    }
}

// ---------------------------------------------------------------------------
// Tiled WMMA GEMM: C[m][n] = sum_k A[m][k] * Bm[n][k]   (A:[M][K], Bm:[N][K], f16)
// Block tile 128x128, 8 waves, each wave 32x64 (2x4 wmma tiles), K-step 32.
// Double-buffered LDS fed by async global->LDS copies; K%64==0 required.
// EPI==0: Aout[m][n] = f16(silu(C + bias[n]))
// EPI==1: atomicAdd(out[m], sum_n silu(C + bias[n]) * w3[n])
// ---------------------------------------------------------------------------
#define LDT 40   // LDS row stride in halves (32 + 8 pad, even -> 4B-aligned pairs)

// One 32-deep K-slab of WMMA work from a fixed pair of LDS tiles.
__device__ __forceinline__ void compute_slab(const _Float16* lA, const _Float16* lB,
                                             int wid, int lane, v8f (&acc)[2][4]) {
    const int hi = lane >> 4;
    // A fragments: lane = M (mod 16); ISA K mapping (hi*8 / 16+hi*8 pairs)
    // -> two contiguous 16B groups per fragment (ds_load_b128 x2).
    v16h afrag[2];
    #pragma unroll
    for (int mt = 0; mt < 2; ++mt) {
        const _Float16* base = lA + ((wid & 3) * 32 + mt * 16 + (lane & 15)) * LDT;
        #pragma unroll
        for (int v = 0; v < 8; ++v) {
            const int k = (v < 4) ? (hi * 8 + 2 * v) : (16 + hi * 8 + 2 * (v - 4));
            const h2 p = *(const h2*)(base + k);
            afrag[mt][2 * v]     = p[0];
            afrag[mt][2 * v + 1] = p[1];
        }
    }
    // B fragments: lane = N (mod 16); same ISA K mapping as A (hi*8 / 16+hi*8 pairs).
    v16h bfrag[4];
    #pragma unroll
    for (int nt = 0; nt < 4; ++nt) {
        const _Float16* base = lB + ((wid >> 2) * 64 + nt * 16 + (lane & 15)) * LDT;
        #pragma unroll
        for (int v = 0; v < 8; ++v) {
            const int k = (v < 4) ? (hi * 8 + 2 * v) : (16 + hi * 8 + 2 * (v - 4));
            const h2 p = *(const h2*)(base + k);
            bfrag[nt][2 * v]     = p[0];
            bfrag[nt][2 * v + 1] = p[1];
        }
    }
    #pragma unroll
    for (int mt = 0; mt < 2; ++mt)
        #pragma unroll
        for (int nt = 0; nt < 4; ++nt)
            acc[mt][nt] = __builtin_amdgcn_wmma_f32_16x16x32_f16(
                false, afrag[mt], false, bfrag[nt],
                (short)0, acc[mt][nt], false, false);
}

template <int EPI>
__global__ __launch_bounds__(256)
void gemm_kernel(const _Float16* __restrict__ A, const _Float16* __restrict__ Bm,
                 const float* __restrict__ bias, const float* __restrict__ w3,
                 _Float16* __restrict__ Aout, float* __restrict__ out,
                 int N, int K) {
    __shared__ _Float16 ldsA[2][128 * LDT];
    __shared__ _Float16 ldsB[2][128 * LDT];

    const int tid    = threadIdx.x;
    const int lane   = tid & 31;
    const int wid    = tid >> 5;
    const int blockN = blockIdx.x * 128;
    const int blockM = blockIdx.y * 128;
    const int mbase  = blockM + (wid & 3) * 32;   // wave's M origin (32 rows)
    const int nbase  = blockN + (wid >> 2) * 64;  // wave's N origin (64 cols)

    // Per-thread staging assignment: two fixed 16B chunks per tile.
    //   chunk0: (row, seg) = (tid>>2, tid&3); chunk1: (row+64, seg)
    const int srow = tid >> 2;
    const int sseg = (tid & 3) * 8;                     // halves
    const _Float16* gA0 = A  + (size_t)(blockM + srow) * K + sseg;
    const _Float16* gA1 = A  + (size_t)(blockM + srow + 64) * K + sseg;
    const _Float16* gB0 = Bm + (size_t)(blockN + srow) * K + sseg;
    const _Float16* gB1 = Bm + (size_t)(blockN + srow + 64) * K + sseg;
    const unsigned lA0[2] = { lds_off32(&ldsA[0][srow * LDT + sseg]),
                              lds_off32(&ldsA[1][srow * LDT + sseg]) };
    const unsigned lA1[2] = { lds_off32(&ldsA[0][(srow + 64) * LDT + sseg]),
                              lds_off32(&ldsA[1][(srow + 64) * LDT + sseg]) };
    const unsigned lB0[2] = { lds_off32(&ldsB[0][srow * LDT + sseg]),
                              lds_off32(&ldsB[1][srow * LDT + sseg]) };
    const unsigned lB1[2] = { lds_off32(&ldsB[0][(srow + 64) * LDT + sseg]),
                              lds_off32(&ldsB[1][(srow + 64) * LDT + sseg]) };

    v8f acc[2][4];
    #pragma unroll
    for (int mt = 0; mt < 2; ++mt)
        #pragma unroll
        for (int nt = 0; nt < 4; ++nt)
            acc[mt][nt] = (v8f){0.f,0.f,0.f,0.f,0.f,0.f,0.f,0.f};

    // Stage K-slab (k0, in halves) into buffer b: 4 unconditional async copies.
    auto stage = [&](int k0, int b) {
        async_copy_b128(lA0[b], gA0 + k0);
        async_copy_b128(lA1[b], gA1 + k0);
        async_copy_b128(lB0[b], gB0 + k0);
        async_copy_b128(lB1[b], gB1 + k0);
    };

    // Prologue: slab 0 -> buffer 0.
    stage(0, 0);
    wait_async0();
    __syncthreads();

    // K % 64 == 0: two slabs per trip so buffer indices are constants.
    for (int k0 = 0; k0 < K; k0 += 64) {
        stage(k0 + 32, 1);                       // prefetch slab k0+32 -> buf1
        compute_slab(ldsA[0], ldsB[0], wid, lane, acc);
        wait_async0();
        __syncthreads();

        if (k0 + 64 < K) stage(k0 + 64, 0);      // prefetch slab k0+64 -> buf0
        compute_slab(ldsA[1], ldsB[1], wid, lane, acc);
        wait_async0();
        __syncthreads();
    }

    // Epilogues. C layout: lanes 0-15 rows M=0..7 (VGPR r), lanes 16-31 rows M=8..15.
    const int mrow0 = mbase + ((lane >> 4) * 8);
    if (EPI == 0) {
        #pragma unroll
        for (int mt = 0; mt < 2; ++mt)
            #pragma unroll
            for (int nt = 0; nt < 4; ++nt) {
                const int n  = nbase + nt * 16 + (lane & 15);
                const float bv = bias[n];
                #pragma unroll
                for (int r = 0; r < 8; ++r) {
                    const float s = silu_f(acc[mt][nt][r] + bv);
                    Aout[(size_t)(mrow0 + mt * 16 + r) * N + n] = (_Float16)s;
                }
            }
    } else {
        #pragma unroll
        for (int mt = 0; mt < 2; ++mt) {
            float sum[8] = {0.f,0.f,0.f,0.f,0.f,0.f,0.f,0.f};
            #pragma unroll
            for (int nt = 0; nt < 4; ++nt) {
                const int n  = nbase + nt * 16 + (lane & 15);
                const float bv = bias[n];
                const float wv = w3[n];
                #pragma unroll
                for (int r = 0; r < 8; ++r)
                    sum[r] += silu_f(acc[mt][nt][r] + bv) * wv;
            }
            #pragma unroll
            for (int r = 0; r < 8; ++r) {
                float v = sum[r];
                #pragma unroll
                for (int ofs = 8; ofs >= 1; ofs >>= 1)
                    v += __shfl_xor(v, ofs, 32);  // reduce 16 lanes sharing row m
                if ((lane & 15) == 0)
                    atomicAdd(out + mrow0 + mt * 16 + r, v);
            }
        }
    }
}

// ---------------------------------------------------------------------------
// Workspace layout (bytes, 256-aligned): xs16 | a1 | W1h | W2h  (~97.5 MB)
// ---------------------------------------------------------------------------
#define XS_OFF   ((size_t)0)
#define A1_OFF   ((size_t)(BB * DD * 2))                     //   1 MB
#define W1H_OFF  (A1_OFF + (size_t)BB * HH * 2)              // + 64 MB
#define W2H_OFF  (W1H_OFF + (size_t)HH * DD * 2)             // + 0.5 MB

extern "C" void kernel_launch(void* const* d_in, const int* in_sizes, int n_in,
                              void* d_out, int out_size, void* d_ws, size_t ws_size,
                              hipStream_t stream) {
    const float* x     = (const float*)d_in[0];
    const float* Wprod = (const float*)d_in[1];
    const float* W1    = (const float*)d_in[2];
    const float* b1    = (const float*)d_in[3];
    const float* W2    = (const float*)d_in[4];
    const float* b2    = (const float*)d_in[5];
    const float* W3    = (const float*)d_in[6];
    const float* b3    = (const float*)d_in[7];
    float* out = (float*)d_out;

    char* ws = (char*)d_ws;
    _Float16* xs  = (_Float16*)(ws + XS_OFF);
    _Float16* a1  = (_Float16*)(ws + A1_OFF);
    _Float16* W1h = (_Float16*)(ws + W1H_OFF);
    _Float16* W2h = (_Float16*)(ws + W2H_OFF);

    // 1) prod_sum + b3 -> out ; f16(silu(x)) -> xs
    prep_kernel<<<BB / 8, 256, 0, stream>>>(x, Wprod, b3, xs, out);
    // 2) weight conversions f32 -> f16
    cvt_kernel<<<(HH * DD / 4 + 255) / 256, 256, 0, stream>>>(W1, W1h, HH * DD / 4);
    cvt_kernel<<<(HH * HH / 4 + 255) / 256, 256, 0, stream>>>(W2, W2h, HH * HH / 4);
    // 3) h1 = silu(xs@W1^T + b1) -> a1 (f16)
    gemm_kernel<0><<<dim3(HH / 128, BB / 128), 256, 0, stream>>>(
        xs, W1h, b1, nullptr, a1, nullptr, HH, DD);
    // 4) out += silu(a1@W2^T + b2) @ W3^T   (GEMM3 fused into epilogue)
    gemm_kernel<1><<<dim3(HH / 128, BB / 128), 256, 0, stream>>>(
        a1, W2h, b2, W3, nullptr, out, HH, HH);
}
